// GNN_Layer_66640712564799
// MI455X (gfx1250) — matrix operand order repaired and checked by reading, not compiled, not run on hardware
//
#include <hip/hip_runtime.h>
#include <hip/hip_bf16.h>

// ---------------------------------------------------------------------------
// GNN layer for MI455X (gfx1250, wave32, WMMA).
// Dominant work: two 6000x6000x128 GEMMs -> bf16 WMMA (16x16x32) with f32
// accumulate, HBM-roofline bound (~38us total traffic @ 23.3 TB/s).
// GEMM: double-buffered LDS with ONE barrier per K-tile; tile k+1 streams
// from HBM into registers while tile k computes from LDS; native packed
// f32->bf16 converts; tile-major pre-transposed B operand so the hot loop
// does only aligned b128 LDS stores. amdgpu_waves_per_eu(2) raises the VGPR
// budget so the staged tile stays in registers (no scratch spills).
// Sparse WR@H parts done as edge-list scatters with f32 global atomics.
// ---------------------------------------------------------------------------

#define ND_    6000
#define NT_    6000
#define FDIM   128
#define PD     16
#define NEDGE  100000
#define KT     32
#define KTILES ((ND_ + KT - 1) / KT)          // 188
#define BTILE  (FDIM * KT)                    // 4096 bf16 elems per B k-tile

typedef __attribute__((ext_vector_type(16))) __bf16          v16bf;
typedef __attribute__((ext_vector_type(8)))  __bf16          v8bf;
typedef __attribute__((ext_vector_type(4)))  __bf16          v4bf;
typedef __attribute__((ext_vector_type(8)))  float           v8f;

// ---------------- degree sums ----------------
__global__ __launch_bounds__(256) void rowsum_kernel(
    const float* __restrict__ M, float* __restrict__ out, int ncols)
{
    __shared__ float red[256];
    const int r = blockIdx.x;
    const float* row = M + (long)r * ncols;
    float s = 0.f;
    for (int c = threadIdx.x; c < ncols; c += 256) s += row[c];
    red[threadIdx.x] = s;
    __syncthreads();
    for (int off = 128; off > 0; off >>= 1) {
        if (threadIdx.x < off) red[threadIdx.x] += red[threadIdx.x + off];
        __syncthreads();
    }
    if (threadIdx.x == 0) out[r] = red[0];
}

__global__ __launch_bounds__(256) void colsum_kernel(
    const float* __restrict__ M, float* __restrict__ out,
    int nrows, int ncols, int rows_per)
{
    const int c = blockIdx.x * 256 + threadIdx.x;
    if (c >= ncols) return;
    const int r0 = blockIdx.y * rows_per;
    int r1 = r0 + rows_per; if (r1 > nrows) r1 = nrows;
    float s = 0.f;
    for (int r = r0; r < r1; ++r) s += M[(long)r * ncols + c];
    atomicAdd(&out[c], s);
}

__global__ __launch_bounds__(256) void invsqrt_kernel(float* v, int n)
{
    const int i = blockIdx.x * 256 + threadIdx.x;
    if (i >= n) return;
    float s = v[i];
    if (s == 0.f) s = 1.f;
    v[i] = rsqrtf(s);
}

// ---------------- edge weights: sigmoid(profile @ W_pro) ----------------
__global__ __launch_bounds__(256) void edge_w_kernel(
    const float* __restrict__ prof, const float* __restrict__ Wp,
    float* __restrict__ w, int n)
{
    const int e = blockIdx.x * 256 + threadIdx.x;
    if (e >= n) return;
    const float* p = prof + (long)e * PD;
    float x = 0.f;
#pragma unroll
    for (int k = 0; k < PD; ++k) x += p[k] * Wp[k];
    w[e] = 1.f / (1.f + __expf(-x));
}

// ---------------- sparse WR @ H_t and WR^T @ H_d via edge scatter ----------
// 32 lanes per edge, 4 columns per lane (float4 gathers, f32 global atomics).
__global__ __launch_bounds__(256) void edge_scatter_kernel(
    const int* __restrict__ idxs, const float* __restrict__ w,
    const float* __restrict__ R,
    const float* __restrict__ dt, const float* __restrict__ td,
    const float* __restrict__ Ht, const float* __restrict__ Hd,
    float* __restrict__ Hdt, float* __restrict__ Htd, int n)
{
    const int gid  = blockIdx.x * 256 + threadIdx.x;
    const int e    = gid >> 5;
    const int lane = gid & 31;
    if (e >= n) return;
    const int i = idxs[2 * e], j = idxs[2 * e + 1];
    const float we = w[e] * dt[i] * R[(long)i * NT_ + j] * td[j];
    const int c = lane * 4;
    const float4 ht = *(const float4*)(Ht + (long)j * FDIM + c);
    const float4 hd = *(const float4*)(Hd + (long)i * FDIM + c);
    float* pdt = Hdt + (long)i * FDIM + c;
    float* ptd = Htd + (long)j * FDIM + c;
    atomicAdd(pdt + 0, we * ht.x); atomicAdd(pdt + 1, we * ht.y);
    atomicAdd(pdt + 2, we * ht.z); atomicAdd(pdt + 3, we * ht.w);
    atomicAdd(ptd + 0, we * hd.x); atomicAdd(ptd + 1, we * hd.y);
    atomicAdd(ptd + 2, we * hd.z); atomicAdd(ptd + 3, we * hd.w);
}

// ---- prescale H by degree, convert to bf16, store tile-major TRANSPOSED ---
// Output HsT: [KTILES][128 cols][32 k] -- each 8KB k-tile contiguous, so the
// GEMM B path needs no transpose and no tail logic (zero-padded here).
__global__ __launch_bounds__(256) void scale_bf16_t_kernel(
    const float* __restrict__ H, const float* __restrict__ sc,
    unsigned short* __restrict__ HsT, int rows)
{
    __shared__ __align__(16) unsigned short tile[KT][136];   // [km][c], padded
    const int kt  = blockIdx.x;
    const int tid = threadIdx.x;
    // load 32 rows x 128 cols of H (coalesced), scale, convert
#pragma unroll
    for (int it = 0; it < 4; ++it) {
        const int slot = tid + 256 * it;        // 0..1023 float4 slots
        const int km = slot >> 5;               // 0..31
        const int c  = (slot & 31) << 2;        // 0..124
        const int k  = kt * KT + km;
        float4 h; h.x = h.y = h.z = h.w = 0.f;
        float s = 0.f;
        if (k < rows) { s = sc[k]; h = *(const float4*)(H + (long)k * FDIM + c); }
        const v4bf o = { (__bf16)(s * h.x), (__bf16)(s * h.y),
                         (__bf16)(s * h.z), (__bf16)(s * h.w) };
        *(v4bf*)&tile[km][c] = o;
    }
    __syncthreads();
    // write transposed, fully contiguous: thread t -> col c = t>>1, k-half kh
    const int c  = tid >> 1;
    const int kh = tid & 1;
    unsigned short u[16];
#pragma unroll
    for (int i = 0; i < 16; ++i) u[i] = tile[kh * 16 + i][c];
    uint4 q0, q1;
    q0.x = (unsigned)u[0]  | ((unsigned)u[1]  << 16);
    q0.y = (unsigned)u[2]  | ((unsigned)u[3]  << 16);
    q0.z = (unsigned)u[4]  | ((unsigned)u[5]  << 16);
    q0.w = (unsigned)u[6]  | ((unsigned)u[7]  << 16);
    q1.x = (unsigned)u[8]  | ((unsigned)u[9]  << 16);
    q1.y = (unsigned)u[10] | ((unsigned)u[11] << 16);
    q1.z = (unsigned)u[12] | ((unsigned)u[13] << 16);
    q1.w = (unsigned)u[14] | ((unsigned)u[15] << 16);
    unsigned short* dst = HsT + (long)kt * BTILE + c * KT + kh * 16;
    *(uint4*)dst       = q0;
    *(uint4*)(dst + 8) = q1;
}

// ---------------- main GEMM: Out = relu(rs_row * (A @ BsT)), bf16 WMMA -----
// A: [N][K] f32 row-major (D or T), converted to bf16 while staging in LDS.
// BsT: tile-major transposed prescaled H. 8 waves/WG; each wave computes a
// 16x128 strip via 8 accumulators of V_WMMA_F32_16X16X32_BF16.
__device__ __forceinline__ void gload_A(
    const float* __restrict__ A, int rBase, int kBase, int N, int K, int tid,
    float4 (&ar)[4])
{
#pragma unroll
    for (int it = 0; it < 4; ++it) {
        const int slot = tid + 256 * it;        // 0..1023 float4 slots
        const int rr = slot >> 3;               // 0..127
        const int kq = (slot & 7) << 2;         // 0..28 step 4
        const int gr = rBase + rr;
        float4 v; v.x = v.y = v.z = v.w = 0.f;
        if (gr < N) {
            const long base = (long)gr * K + kBase + kq;
            if (kBase + kq + 3 < K) {
                v = *(const float4*)(A + base);
                if (it == 0 && kBase + 2 * KT < K)      // warm L2 two tiles ahead
                    __builtin_prefetch(A + base + 2 * KT, 0, 0);
            } else {
                float t[4] = {0.f, 0.f, 0.f, 0.f};
                for (int q = 0; q < 4; ++q)
                    if (kBase + kq + q < K) t[q] = A[base + q];
                v.x = t[0]; v.y = t[1]; v.z = t[2]; v.w = t[3];
            }
        }
        ar[it] = v;
    }
}

__device__ __forceinline__ void gload_B(
    const unsigned short* __restrict__ BsT, int kBase, int tid, uint4 (&br)[2])
{
    // one contiguous 8KB tile; thread t covers elems [t*16, t*16+16)
    const unsigned short* p = BsT + (long)(kBase / KT) * BTILE + tid * 16;
    br[0] = *(const uint4*)p;
    br[1] = *(const uint4*)(p + 8);
}

__device__ __forceinline__ void commit_tiles(
    unsigned short (*la)[40], unsigned short (*lb)[40], int tid,
    const float4 (&ar)[4], const uint4 (&br)[2])
{
#pragma unroll
    for (int it = 0; it < 4; ++it) {
        const int slot = tid + 256 * it;
        const int rr = slot >> 3;
        const int kq = (slot & 7) << 2;
        const v4bf o = { (__bf16)ar[it].x, (__bf16)ar[it].y,
                         (__bf16)ar[it].z, (__bf16)ar[it].w };
        *(v4bf*)&la[rr][kq] = o;
    }
    *(uint4*)&lb[tid >> 1][(tid & 1) * 16]     = br[0];
    *(uint4*)&lb[tid >> 1][(tid & 1) * 16 + 8] = br[1];
}

__global__ __launch_bounds__(256)
__attribute__((amdgpu_waves_per_eu(2)))
void wmma_gemm_relu_kernel(
    const float* __restrict__ A, const unsigned short* __restrict__ BsT,
    const float* __restrict__ rs, float* __restrict__ Out, int N, int K)
{
    __shared__ __align__(16) unsigned short ldsA[2][128][40];  // [buf][row][k]
    __shared__ __align__(16) unsigned short ldsB[2][128][40];  // [buf][n][k]

    const int tid  = threadIdx.x;
    const int wave = tid >> 5;          // wave32
    const int lane = tid & 31;
    const int half = lane >> 4;
    const int lr   = lane & 15;
    const int rBase = blockIdx.x * 128;

    v8f acc[8];
    const v8f z = {};
#pragma unroll
    for (int c = 0; c < 8; ++c) acc[c] = z;

    // prologue: tile 0 -> buf 0
    float4 aReg[4];
    uint4  bReg[2];
    gload_A(A, rBase, 0, N, K, tid, aReg);
    gload_B(BsT, 0, tid, bReg);
    commit_tiles(ldsA[0], ldsB[0], tid, aReg, bReg);
    __syncthreads();

    int p = 0;
    for (int kBase = 0; kBase < K; kBase += KT) {
        const bool more = (kBase + KT) < K;
        // ---- stream tile k+1 from HBM while computing tile k ----
        if (more) {
            gload_A(A, rBase, kBase + KT, N, K, tid, aReg);
            gload_B(BsT, kBase + KT, tid, bReg);
        }

        // ---- A fragment: lanes 0-15 -> K{0..7,16..23}, 16-31 -> K{8..15,24..31}
        const int arow = wave * 16 + lr;
        const v8bf alo = *(const v8bf*)&ldsA[p][arow][half * 8];
        const v8bf ahi = *(const v8bf*)&ldsA[p][arow][16 + half * 8];
        v16bf afrag;
#pragma unroll
        for (int q = 0; q < 8; ++q) { afrag[q] = alo[q]; afrag[8 + q] = ahi[q]; }

        // ---- B fragments loaded per block; scheduler pipelines the ds reads
#pragma unroll
        for (int c = 0; c < 8; ++c) {
            const int n = c * 16 + lr;      // lane = column within 16-col block
            const v8bf blo = *(const v8bf*)&ldsB[p][n][half * 8];
            const v8bf bhi = *(const v8bf*)&ldsB[p][n][16 + half * 8];
            v16bf bfrag;
#pragma unroll
            for (int q = 0; q < 8; ++q) { bfrag[q] = blo[q]; bfrag[8 + q] = bhi[q]; }
            acc[c] = __builtin_amdgcn_wmma_f32_16x16x32_bf16(
                false, afrag, false, bfrag, (short)0, acc[c], false, false);
        }

        // ---- commit tile k+1 into the OTHER buffer (no barrier needed
        //      before the store: same-wave LDS ops are in-order, other waves
        //      are protected by the single barrier below) ----
        if (more) commit_tiles(ldsA[p ^ 1], ldsB[p ^ 1], tid, aReg, bReg);
        __syncthreads();
        p ^= 1;
    }

    // ---- epilogue: row scale + relu ----
    float dsc[8];
    int rows_[8];
#pragma unroll
    for (int v = 0; v < 8; ++v) {
        const int row = rBase + wave * 16 + v + 8 * half;
        rows_[v] = row;
        dsc[v] = (row < N) ? rs[row] : 0.f;
    }
#pragma unroll
    for (int c = 0; c < 8; ++c) {
        const int col = c * 16 + lr;
#pragma unroll
        for (int v = 0; v < 8; ++v) {
            if (rows_[v] < N)
                Out[(long)rows_[v] * FDIM + col] = fmaxf(dsc[v] * acc[c][v], 0.f);
        }
    }
}

// ---------------- final dense: relu(concat[H, relu(Hxy), Hxx] @ W) ---------
__global__ __launch_bounds__(256) void dense_out_kernel(
    const float* __restrict__ Hraw, const float* __restrict__ Hxy,
    const float* __restrict__ Hxx,  const float* __restrict__ W,
    float* __restrict__ Out, int rows)
{
    const int id = blockIdx.x * 256 + threadIdx.x;
    if (id >= rows * FDIM) return;
    const int r = id >> 7, u = id & 127;
    const float* h0 = Hraw + (long)r * FDIM;
    const float* h1 = Hxy  + (long)r * FDIM;
    const float* h2 = Hxx  + (long)r * FDIM;
    float acc = 0.f;
    for (int k = 0; k < FDIM; ++k) acc += h0[k] * W[k * FDIM + u];
    for (int k = 0; k < FDIM; ++k) acc += fmaxf(h1[k], 0.f) * W[(FDIM + k) * FDIM + u];
    for (int k = 0; k < FDIM; ++k) acc += h2[k] * W[(2 * FDIM + k) * FDIM + u];
    Out[id] = fmaxf(acc, 0.f);
}

// ---------------------------------------------------------------------------
extern "C" void kernel_launch(void* const* d_in, const int* in_sizes, int n_in,
                              void* d_out, int out_size, void* d_ws, size_t ws_size,
                              hipStream_t stream)
{
    const float* R   = (const float*)d_in[0];
    const float* D   = (const float*)d_in[1];
    const float* T   = (const float*)d_in[2];
    const float* Hd  = (const float*)d_in[3];
    const float* Ht  = (const float*)d_in[4];
    const float* pp  = (const float*)d_in[5];
    const float* np  = (const float*)d_in[6];
    const int*   pidx = (const int*)d_in[7];
    const int*   nidx = (const int*)d_in[8];
    const float* Wpro = (const float*)d_in[9];
    const float* Wdd  = (const float*)d_in[10];
    const float* Wdt  = (const float*)d_in[11];
    float* out = (float*)d_out;

    // workspace layout (f32 unless noted) ~16.3 MB
    float* dt  = (float*)d_ws;            // [6000]
    float* td  = dt  + ND_;               // [6000]
    float* dd  = td  + NT_;               // [6000]
    float* tt  = dd  + ND_;               // [6000]
    float* pw  = tt  + NT_;               // [100000]
    float* nw  = pw  + NEDGE;             // [100000]
    float* Hdt = nw  + NEDGE;             // [6000*128]
    float* Htd = Hdt + ND_ * FDIM;        // [6000*128]
    float* Hdd = Htd + NT_ * FDIM;        // [6000*128]
    float* Htt = Hdd + ND_ * FDIM;        // [6000*128]
    unsigned short* HsD = (unsigned short*)(Htt + NT_ * FDIM);  // bf16 tile-major
    unsigned short* HsT = HsD + (size_t)KTILES * BTILE;

    // 1) degree sums (td accumulated via atomics -> zero first)
    hipMemsetAsync(td, 0, NT_ * sizeof(float), stream);
    rowsum_kernel<<<ND_, 256, 0, stream>>>(R, dt, NT_);
    {
        const int ychunks = 16;
        const int rows_per = (ND_ + ychunks - 1) / ychunks;
        colsum_kernel<<<dim3((NT_ + 255) / 256, ychunks), 256, 0, stream>>>(
            R, td, ND_, NT_, rows_per);
    }
    rowsum_kernel<<<ND_, 256, 0, stream>>>(D, dd, ND_);
    rowsum_kernel<<<NT_, 256, 0, stream>>>(T, tt, NT_);
    // dt/td/dd/tt are contiguous: one fused invsqrt pass
    invsqrt_kernel<<<(4 * ND_ + 255) / 256, 256, 0, stream>>>(dt, 4 * ND_);

    // 2) learned edge weights
    edge_w_kernel<<<(NEDGE + 255) / 256, 256, 0, stream>>>(pp, Wpro, pw, NEDGE);
    edge_w_kernel<<<(NEDGE + 255) / 256, 256, 0, stream>>>(np, Wpro, nw, NEDGE);

    // 3) sparse WR @ H_t / WR^T @ H_d via edge scatter
    hipMemsetAsync(Hdt, 0, 2 * (size_t)ND_ * FDIM * sizeof(float), stream);
    {
        const int g = (NEDGE * 32 + 255) / 256;
        edge_scatter_kernel<<<g, 256, 0, stream>>>(pidx, pw, R, dt, td, Ht, Hd, Hdt, Htd, NEDGE);
        edge_scatter_kernel<<<g, 256, 0, stream>>>(nidx, nw, R, dt, td, Ht, Hd, Hdt, Htd, NEDGE);
    }

    // 4) degree-prescale H, convert to bf16, tile-major transpose
    scale_bf16_t_kernel<<<KTILES, 256, 0, stream>>>(Hd, dd, HsD, ND_);
    scale_bf16_t_kernel<<<KTILES, 256, 0, stream>>>(Ht, tt, HsT, NT_);

    // 5) the two big GEMMs: H_dd = relu(dd*(D @ (dd*H_d))), same for T
    {
        const int g = (ND_ + 127) / 128;  // 47
        wmma_gemm_relu_kernel<<<g, 256, 0, stream>>>(D, HsD, dd, Hdd, ND_, ND_);
        wmma_gemm_relu_kernel<<<g, 256, 0, stream>>>(T, HsT, tt, Htt, NT_, NT_);
    }

    // 6) final dense layers (relu applied to Hdt/Htd on read)
    {
        const int g = (ND_ * FDIM + 255) / 256;
        dense_out_kernel<<<g, 256, 0, stream>>>(Hd, Hdt, Hdd, Wdd, out, ND_);
        dense_out_kernel<<<g, 256, 0, stream>>>(Ht, Htd, Htt, Wdt, out + (size_t)ND_ * FDIM, NT_);
    }
}
